// MultiHeadAttention_10780367913274
// MI455X (gfx1250) — matrix operand order repaired and checked by reading, not compiled, not run on hardware
//
#include <hip/hip_runtime.h>
#include <hip/hip_bf16.h>

// ---------------------------------------------------------------------------
// MHA forward for MI455X (gfx1250, wave32, WMMA 16x16x32 bf16)
// B=4, S=2048, D_MODEL=1024, H=16, d_k=64
// d_out = [output f32 (8,388,608)] ++ [attn f32 (268,435,456)]
// Roofline: ~138 GFLOP, but 1 GiB attn write dominates (~46us @ 23.3 TB/s)
// -> fuse scores+softmax+attn@V in LDS (320 KB/WGP), NT-store the attn stream.
// ---------------------------------------------------------------------------

typedef __attribute__((ext_vector_type(16))) __bf16        v16bf;
typedef __attribute__((ext_vector_type(8)))  float         v8f;
typedef __attribute__((ext_vector_type(4)))  float         v4f;
typedef __attribute__((ext_vector_type(8)))  unsigned int  v8u;
typedef __attribute__((ext_vector_type(4)))  unsigned int  v4u;
typedef __attribute__((ext_vector_type(2)))  unsigned int  v2u;
typedef __attribute__((ext_vector_type(4)))  int           v4i;

typedef unsigned int   u32;
typedef unsigned short u16;

#define B_DIM   4
#define S_LEN   2048
#define D_MODEL 1024
#define H_NUM   16
#define D_K     64
#define QT      16            // q-rows per workgroup in attention

// Workspace layout (bytes) -- total 120 MiB
#define XQ_OFF  0u            // 16 MiB  bf16 query
#define XK_OFF  16777216u     // 16 MiB  bf16 key
#define XV_OFF  33554432u     // 16 MiB  bf16 value
#define WQ_OFF  50331648u     //  2 MiB  bf16 w_q^T [N,K]
#define WK_OFF  52428800u
#define WV_OFF  54525952u
#define WO_OFF  56623104u
#define QB_OFF  58720256u     // 16 MiB  bf16 Q  [B*S, D]
#define KB_OFF  75497472u     // 16 MiB  bf16 K  [B*S, D]
#define VT_OFF  92274688u     // 16 MiB  bf16 V^T [B,H,64,S]
#define CTX_OFF 109051904u    // 16 MiB  bf16 context [B*S, D]

// Attention LDS regions (dynamic)
#define LDS_SC  0u            // f32 scores [16][2048]  = 131072 B
#define LDS_AT  131072u       // bf16 attn  [16][2048]  =  65536 B
#define LDS_RED 196608u       // f32 red    [128]       =    512 B
#define LDS_SQ  197120u       // bf16 Qtile [16][64]    =   2048 B
#define LDS_TOT 199168u

#if defined(__has_builtin)
#if __has_builtin(__builtin_amdgcn_global_load_async_to_lds_b128)
#define HAVE_ASYNC_LDS 1
#endif
#endif

// Address-space-qualified int4 for the async-to-LDS builtin
// (param 0 is int4 in AS1 per the compiler's own diagnostic).
typedef __attribute__((address_space(1))) v4i as1_v4i;
typedef __attribute__((address_space(3))) v4i as3_v4i;

__device__ __forceinline__ u16 f32_to_bf16(float f) {
    u32 u = __builtin_bit_cast(u32, f);
    u32 r = u + 0x7FFFu + ((u >> 16) & 1u);   // round-to-nearest-even
    return (u16)(r >> 16);
}
__device__ __forceinline__ u32 pack_bf16x2(float a, float b) {
    return (u32)f32_to_bf16(a) | ((u32)f32_to_bf16(b) << 16);
}

// A-matrix (16x32 bf16) fragment: lane row = lane&15, half = lane>>4.
// VGPR g holds K pair starting at (g<4 ? 2g : 8+2g) + 8*half.  (ISA 7.12.2)
__device__ __forceinline__ int kA(int g, int hf) {
    return (g < 4 ? 2 * g : 8 + 2 * g) + 8 * hf;
}
// B-matrix (32x16 bf16) fragment: lane col = lane&15, half = lane>>4.
__device__ __forceinline__ int kB(int g, int hf) {
    return (g < 4 ? 2 * g : 8 + 2 * g) + 8 * hf;
}

__device__ __forceinline__ v16bf frag_a(const u16* lane_base, int hf) {
    v8u r;
#pragma unroll
    for (int g = 0; g < 8; ++g)
        r[g] = *(const u32*)(lane_base + kA(g, hf));
    return __builtin_bit_cast(v16bf, r);
}
__device__ __forceinline__ v16bf frag_b(const u16* lane_base, int hf) {
    v8u r;
#pragma unroll
    for (int g = 0; g < 8; ++g)
        r[g] = *(const u32*)(lane_base + kB(g, hf));
    return __builtin_bit_cast(v16bf, r);
}

__device__ __forceinline__ v8f wmma_bf16(v16bf a, v16bf b, v8f c) {
    return __builtin_amdgcn_wmma_f32_16x16x32_bf16(
        false, a, false, b, (short)0, c, false, false);
}

// ---------------------------------------------------------------------------
// Elementwise f32 -> bf16, 4 elems/thread, NT loads (read-once stream)
__global__ void cvt_bf16_kernel(const float* __restrict__ src,
                                u16* __restrict__ dst, int n4) {
    int i = blockIdx.x * 256 + threadIdx.x;
    if (i < n4) {
        v4f s = __builtin_nontemporal_load((const v4f*)src + i);
        v2u p;
        p[0] = pack_bf16x2(s.x, s.y);
        p[1] = pack_bf16x2(s.z, s.w);
        *((v2u*)dst + i) = p;
    }
}

// Weight transpose+convert: w[K,N] f32 -> wt[N,K] bf16 (coalesced writes)
__global__ void cvt_wt_kernel(const float* __restrict__ w,
                              u16* __restrict__ wt) {
    int i  = blockIdx.x * 256 + threadIdx.x;   // i = nn*1024 + kk
    int nn = i >> 10, kk = i & 1023;
    wt[i] = f32_to_bf16(w[(size_t)kk * D_MODEL + nn]);
}

// ---------------------------------------------------------------------------
// GEMM: C[M,1024] = X[M,1024] @ W + bias, W supplied transposed WT[N,K] bf16.
// Workgroup = 64x64 tile, 4 waves; wave w -> rows [64*bx+16w, +16), 4 n-tiles.
// MODE 0: bf16 out row-major   MODE 1: bf16 out V^T [B,H,64,S]   MODE 2: f32 NT
template <int MODE>
__global__ void gemm_kernel(const u16* __restrict__ X,
                            const u16* __restrict__ WT,
                            const float* __restrict__ bias,
                            void* __restrict__ out) {
    const int tid  = threadIdx.x;
    const int wave = tid >> 5, lane = tid & 31;
    const int m    = lane & 15, hf = lane >> 4;
    const int m0   = blockIdx.x * 64 + wave * 16;
    const int n0   = blockIdx.y * 64;

    v8f acc[4];
#pragma unroll
    for (int nt = 0; nt < 4; ++nt) acc[nt] = (v8f){};

    const u16* Xrow = X + (size_t)(m0 + m) * D_MODEL;

    for (int k0 = 0; k0 < D_MODEL; k0 += 32) {
        __builtin_prefetch(Xrow + k0 + 256, 0, 0);
        v16bf a = frag_a(Xrow + k0, hf);
#pragma unroll
        for (int nt = 0; nt < 4; ++nt) {
            const u16* wl = WT + (size_t)(n0 + nt * 16 + m) * D_MODEL + k0;
            v16bf bfr = frag_b(wl, hf);
            acc[nt] = wmma_bf16(a, bfr, acc[nt]);
        }
    }

#pragma unroll
    for (int nt = 0; nt < 4; ++nt) {
        const int col = n0 + nt * 16 + m;
        const float bv = bias[col];
#pragma unroll
        for (int g = 0; g < 8; ++g) {
            const int row = m0 + g + 8 * hf;
            const float v = acc[nt][g] + bv;
            if (MODE == 0) {
                ((u16*)out)[(size_t)row * D_MODEL + col] = f32_to_bf16(v);
            } else if (MODE == 1) {
                const int bb = row >> 11, ss = row & (S_LEN - 1);
                const int hh = col >> 6,  cc = col & (D_K - 1);
                ((u16*)out)[(((size_t)bb * H_NUM + hh) * D_K + cc) * S_LEN + ss]
                    = f32_to_bf16(v);
            } else {
                __builtin_nontemporal_store(
                    v, (float*)out + (size_t)row * D_MODEL + col);
            }
        }
    }
}

// ---------------------------------------------------------------------------
// Fused attention: one WG (4 waves / 128 thr) per (b, h, 16-row q tile).
__global__ void attn_kernel(const u16* __restrict__ Qb,
                            const u16* __restrict__ Kb,
                            const u16* __restrict__ Vt,
                            float* __restrict__ attn_out,
                            u16* __restrict__ Ctx,
                            float scale) {
    extern __shared__ char smem[];
    float* s_sc  = (float*)(smem + LDS_SC);
    u16*   s_at  = (u16*)(smem + LDS_AT);
    float* s_red = (float*)(smem + LDS_RED);
    u16*   s_q   = (u16*)(smem + LDS_SQ);

    const int tid  = threadIdx.x;
    const int wave = tid >> 5, lane = tid & 31;
    const int m    = lane & 15, hf = lane >> 4;
    const int qt = blockIdx.x, h = blockIdx.y, bz = blockIdx.z;

    const size_t rowQ0 = (size_t)bz * S_LEN + qt * QT;

    // ---- Stage Q tile [16][64] bf16 into LDS (async copy, 16 B per thread)
    {
        const int qr = tid >> 3, qp = tid & 7;        // row 0..15, 16B chunk 0..7
        const u16* gsrc = Qb + (rowQ0 + qr) * D_MODEL + h * D_K + qp * 8;
        u16* ldst = s_q + tid * 8;
#ifdef HAVE_ASYNC_LDS
        __builtin_amdgcn_global_load_async_to_lds_b128(
            (as1_v4i*)(uintptr_t)gsrc,
            (as3_v4i*)(unsigned)(uintptr_t)ldst, 0, 0);
        asm volatile("s_wait_asynccnt 0" ::: "memory");
#else
        *(v4u*)ldst = *(const v4u*)gsrc;
#endif
    }
    __syncthreads();

    const u16* Qlane = s_q + m * D_K;
    const v16bf aq0 = frag_a(Qlane,      hf);   // k = 0..31
    const v16bf aq1 = frag_a(Qlane + 32, hf);   // k = 32..63

    // ---- Phase 1: scores = scale * Q K^T into LDS; waves split kv axis
    const int kv_beg = wave * (S_LEN / 4);
    const int kv_end = kv_beg + (S_LEN / 4);
    for (int kv0 = kv_beg; kv0 < kv_end; kv0 += 16) {
        const u16* Kl = Kb + ((size_t)bz * S_LEN + kv0 + m) * D_MODEL + h * D_K;
        __builtin_prefetch(Kl + 16 * D_MODEL, 0, 0);   // next K tile
        v16bf b0 = frag_b(Kl,      hf);
        v16bf b1 = frag_b(Kl + 32, hf);
        v8f c = (v8f){};
        c = wmma_bf16(aq0, b0, c);
        c = wmma_bf16(aq1, b1, c);
#pragma unroll
        for (int g = 0; g < 8; ++g)
            s_sc[(g + 8 * hf) * S_LEN + kv0 + m] = c[g] * scale;
    }
    __syncthreads();

    // ---- Phase 2: row softmax (8 threads per row, 256 cols each), float4
    const int r = tid >> 3, seg = tid & 7;
    v4f* rowv = (v4f*)(s_sc + r * S_LEN + seg * 256);

    float lmax = -1e30f;
#pragma unroll 4
    for (int i = 0; i < 64; ++i) {
        v4f t = rowv[i];
        lmax = fmaxf(lmax, fmaxf(fmaxf(t.x, t.y), fmaxf(t.z, t.w)));
    }
    s_red[tid] = lmax;
    __syncthreads();
    float rmax = s_red[r * 8];
#pragma unroll
    for (int i = 1; i < 8; ++i) rmax = fmaxf(rmax, s_red[r * 8 + i]);
    __syncthreads();

    float lsum = 0.f;
#pragma unroll 4
    for (int i = 0; i < 64; ++i) {
        v4f t = rowv[i];
        t.x = __expf(t.x - rmax);
        t.y = __expf(t.y - rmax);
        t.z = __expf(t.z - rmax);
        t.w = __expf(t.w - rmax);
        rowv[i] = t;
        lsum += (t.x + t.y) + (t.z + t.w);
    }
    s_red[tid] = lsum;
    __syncthreads();
    float rsum = 0.f;
#pragma unroll
    for (int i = 0; i < 8; ++i) rsum += s_red[r * 8 + i];
    const float inv = 1.f / rsum;

    {   // normalize: NT 128-bit stores of attn (1 GiB write-once stream),
        // packed bf16 copy stays in LDS for the attn@V WMMA.
        const size_t gbase =
            (((size_t)(bz * H_NUM + h) * S_LEN) + qt * QT + r) * S_LEN + seg * 256;
        v4f* gout = (v4f*)(attn_out + gbase);
        u16* abase = s_at + r * S_LEN + seg * 256;
#pragma unroll 4
        for (int i = 0; i < 64; ++i) {
            v4f t = rowv[i];
            t.x *= inv; t.y *= inv; t.z *= inv; t.w *= inv;
            __builtin_nontemporal_store(t, gout + i);
            *(u32*)(abase + 4 * i)     = pack_bf16x2(t.x, t.y);
            *(u32*)(abase + 4 * i + 2) = pack_bf16x2(t.z, t.w);
        }
    }
    __syncthreads();

    // ---- Phase 3: context = attn @ V from LDS (V pre-transposed per head)
    v8f acc[4];
#pragma unroll
    for (int nt = 0; nt < 4; ++nt) acc[nt] = (v8f){};
    const u16* Vhead = Vt + ((size_t)(bz * H_NUM + h) * D_K) * S_LEN;

    for (int kv0 = kv_beg; kv0 < kv_end; kv0 += 32) {
        v16bf aa = frag_a(s_at + (size_t)m * S_LEN + kv0, hf);
#pragma unroll
        for (int nt = 0; nt < 4; ++nt) {
            const u16* vl = Vhead + (size_t)(nt * 16 + m) * S_LEN + kv0;
            v16bf bb = frag_b(vl, hf);
            acc[nt] = wmma_bf16(aa, bb, acc[nt]);
        }
    }

    // cross-wave reduce of 16x64 partials (reuse score LDS region)
    float* cp = (float*)smem;                 // [4][16][64]
#pragma unroll
    for (int nt = 0; nt < 4; ++nt)
#pragma unroll
        for (int g = 0; g < 8; ++g)
            cp[wave * 1024 + (g + 8 * hf) * D_K + nt * 16 + m] = acc[nt][g];
    __syncthreads();

    for (int i = tid; i < QT * D_K; i += 128) {
        float v = cp[i] + cp[1024 + i] + cp[2048 + i] + cp[3072 + i];
        int mm = i >> 6, cc = i & (D_K - 1);
        Ctx[(rowQ0 + mm) * D_MODEL + h * D_K + cc] = f32_to_bf16(v);
    }
}

// ---------------------------------------------------------------------------
extern "C" void kernel_launch(void* const* d_in, const int* in_sizes, int n_in,
                              void* d_out, int out_size, void* d_ws, size_t ws_size,
                              hipStream_t stream) {
    const float* q   = (const float*)d_in[0];
    const float* k   = (const float*)d_in[1];
    const float* v   = (const float*)d_in[2];
    const float* w_q = (const float*)d_in[3];
    const float* b_q = (const float*)d_in[4];
    const float* w_k = (const float*)d_in[5];
    const float* b_k = (const float*)d_in[6];
    const float* w_v = (const float*)d_in[7];
    const float* b_v = (const float*)d_in[8];
    const float* w_o = (const float*)d_in[9];
    const float* b_o = (const float*)d_in[10];

    char* ws = (char*)d_ws;
    u16* Xq  = (u16*)(ws + XQ_OFF);
    u16* Xk  = (u16*)(ws + XK_OFF);
    u16* Xv  = (u16*)(ws + XV_OFF);
    u16* WqT = (u16*)(ws + WQ_OFF);
    u16* WkT = (u16*)(ws + WK_OFF);
    u16* WvT = (u16*)(ws + WV_OFF);
    u16* WoT = (u16*)(ws + WO_OFF);
    u16* Qb  = (u16*)(ws + QB_OFF);
    u16* Kb  = (u16*)(ws + KB_OFF);
    u16* Vt  = (u16*)(ws + VT_OFF);
    u16* Ctx = (u16*)(ws + CTX_OFF);

    float* outp = (float*)d_out;
    float* attn = outp + (size_t)B_DIM * S_LEN * D_MODEL;   // +8,388,608

    const int nAct4 = B_DIM * S_LEN * D_MODEL / 4;          // 2,097,152
    const int nW    = D_MODEL * D_MODEL;                    // 1,048,576

    // bf16 conversions (vectorized x4, NT loads)
    cvt_bf16_kernel<<<nAct4 / 256, 256, 0, stream>>>(q, Xq, nAct4);
    cvt_bf16_kernel<<<nAct4 / 256, 256, 0, stream>>>(k, Xk, nAct4);
    cvt_bf16_kernel<<<nAct4 / 256, 256, 0, stream>>>(v, Xv, nAct4);
    cvt_wt_kernel<<<nW / 256, 256, 0, stream>>>(w_q, WqT);
    cvt_wt_kernel<<<nW / 256, 256, 0, stream>>>(w_k, WkT);
    cvt_wt_kernel<<<nW / 256, 256, 0, stream>>>(w_v, WvT);
    cvt_wt_kernel<<<nW / 256, 256, 0, stream>>>(w_o, WoT);

    // QKV projections (M=8192, N=1024, K=1024)
    dim3 ggrid(128, 16);
    gemm_kernel<0><<<ggrid, 128, 0, stream>>>(Xq, WqT, b_q, Qb);
    gemm_kernel<0><<<ggrid, 128, 0, stream>>>(Xk, WkT, b_k, Kb);
    gemm_kernel<1><<<ggrid, 128, 0, stream>>>(Xv, WvT, b_v, Vt);

    // fused scores + softmax + attn@V
    dim3 agrid(S_LEN / QT, H_NUM, B_DIM);
    attn_kernel<<<agrid, 128, LDS_TOT, stream>>>(Qb, Kb, Vt, attn, Ctx, 0.125f);

    // output projection -> f32 d_out (NT stores)
    gemm_kernel<2><<<ggrid, 128, 0, stream>>>(Ctx, WoT, b_o, outp);
}
